// LSTM_cell_test_17179869186
// MI455X (gfx1250) — matrix operand order repaired and checked by reading, not compiled, not run on hardware
//
#include <hip/hip_runtime.h>
#include <hip/hip_bf16.h>

// ---- types matching gfx1250 WMMA builtin signatures ----
typedef __attribute__((ext_vector_type(16))) __bf16 v16bf;
typedef __attribute__((ext_vector_type(8)))  __bf16 v8bf;
typedef __attribute__((ext_vector_type(4)))  __bf16 v4bf;
typedef __attribute__((ext_vector_type(8)))  float  v8f;

#define IN_DIM   256
#define UNITS    256
#define KK       (IN_DIM + UNITS)         // 512 combined K
#define NCOLS    (4 * UNITS)              // 1024 gate columns
#define KSTEPS   (KK / 32)                // 16 wmma k-steps
#define NTILES   (NCOLS / 16)             // 64 column tiles
#define FRAG_ELEMS 512                    // 32 lanes * 16 bf16 per B fragment
#define MTILES   4                        // 16-row A tiles per wave
#define ROWS_PER_BLOCK (16 * MTILES)      // 64 rows per block
#define LDS_STRIDE (KK + 8)               // bank-shift rows

// Repack [W;U] (fp32, row-major [512 x 1024]) into bf16 B-fragments in ws.
// Fragment f = n_tile*KSTEPS + kb. Within fragment: lane L stores 16
// contiguous bf16 = B[k = kb*32 + (L<16?0:16) + e][col = n_tile*16 + L%16].
__global__ __launch_bounds__(256) void pack_weights_bf16(
    const float* __restrict__ W, const float* __restrict__ U,
    __bf16* __restrict__ ws)
{
    int t = blockIdx.x * blockDim.x + threadIdx.x;      // one thread per (frag,lane)
    if (t >= NTILES * KSTEPS * 32) return;
    int lane   = t & 31;
    int f      = t >> 5;
    int kb     = f & (KSTEPS - 1);
    int n_tile = f / KSTEPS;
    int col    = n_tile * 16 + (lane & 15);
    int kbase  = kb * 32 + ((lane & 16) ? 16 : 0);
    __bf16* dst = ws + (size_t)f * FRAG_ELEMS + (size_t)lane * 16;
#pragma unroll
    for (int e = 0; e < 16; ++e) {
        int kk  = kbase + e;
        float v = (kk < IN_DIM) ? W[(size_t)kk * NCOLS + col]
                                : U[(size_t)(kk - IN_DIM) * NCOLS + col];
        dst[e] = (__bf16)v;
    }
}

__device__ __forceinline__ float sigm(float z) {
    return 1.0f / (1.0f + __expf(-z));
}

__device__ __forceinline__ v16bf cat16(v8bf lo, v8bf hi) {
    return __builtin_shufflevector(lo, hi,
            0,1,2,3,4,5,6,7, 8,9,10,11,12,13,14,15);
}

// Block: 256 threads = 8 waves. Block handles 64 batch rows x 128 units.
// Wave w -> unit tile u_tile = blockIdx.y*8 + w (16 units).
// Each wave: 4 row tiles x 4 gates = 16 accumulators; every B fragment
// fetched from L2 feeds 4 WMMAs (L2-bandwidth + latency amortization).
__global__ __launch_bounds__(256) void lstm_wmma_bf16(
    const float* __restrict__ x, const float* __restrict__ h,
    const float* __restrict__ c, const float* __restrict__ bias,
    const __bf16* __restrict__ wB, float* __restrict__ out, int batch)
{
    __shared__ __align__(16) __bf16 ldsA[ROWS_PER_BLOCK][LDS_STRIDE];

    const int tid     = threadIdx.x;
    const int rowBase = blockIdx.x * ROWS_PER_BLOCK;

    // ---- stage x (K 0..255) and h (K 256..511) for 64 rows as bf16 into LDS ----
    const float4* x4 = (const float4*)x;
    const float4* h4 = (const float4*)h;
#pragma unroll
    for (int i = 0; i < 16; ++i) {
        int idx = tid + i * 256;           // float4 index over [64 rows x 64]
        int row = idx >> 6;
        int c4  = idx & 63;
        float4 vx = x4[(size_t)(rowBase + row) * (IN_DIM / 4) + c4];
        float4 vh = h4[(size_t)(rowBase + row) * (UNITS / 4) + c4];
        v4bf px = { (__bf16)vx.x, (__bf16)vx.y, (__bf16)vx.z, (__bf16)vx.w };
        v4bf ph = { (__bf16)vh.x, (__bf16)vh.y, (__bf16)vh.z, (__bf16)vh.w };
        *(v4bf*)&ldsA[row][c4 * 4]           = px;
        *(v4bf*)&ldsA[row][IN_DIM + c4 * 4]  = ph;
    }
    __syncthreads();

    const int lane   = tid & 31;
    const int wave   = tid >> 5;
    const int u_tile = blockIdx.y * 8 + wave;           // 0..15
    const int arow   = lane & 15;
    const int aoff   = (lane & 16) ? 16 : 0;            // K-half select for A frags

    v8f acc[4][MTILES];                                 // [gate][row tile]
#pragma unroll
    for (int g = 0; g < 4; ++g)
#pragma unroll
        for (int t = 0; t < MTILES; ++t)
            acc[g][t] = (v8f){};

    // Per-gate B fragment streams (advance 512 elems = 1 KB per k-step).
    const __bf16* bp[4];
#pragma unroll
    for (int g = 0; g < 4; ++g)
        bp[g] = wB + ((size_t)(u_tile + g * 16) * KSTEPS) * FRAG_ELEMS
                   + (size_t)lane * 16;

#pragma unroll
    for (int kb = 0; kb < KSTEPS; ++kb) {
        // B fragments for this k-step: 4 coalesced 32B loads (in-order completion)
        v16bf b[4];
#pragma unroll
        for (int g = 0; g < 4; ++g)
            b[g] = *(const v16bf*)(bp[g] + (size_t)kb * FRAG_ELEMS);

        // Four A fragments (row tiles) from LDS
        v16bf a[MTILES];
#pragma unroll
        for (int t = 0; t < MTILES; ++t) {
            const __bf16* ap = &ldsA[t * 16 + arow][kb * 32 + aoff];
            a[t] = cat16(*(const v8bf*)ap, *(const v8bf*)(ap + 8));
        }

        // Consume each B fragment 4x, in load order -> partial waits,
        // 4 back-to-back WMMAs per exposed wait.
#pragma unroll
        for (int g = 0; g < 4; ++g)
#pragma unroll
            for (int t = 0; t < MTILES; ++t)
                acc[g][t] = __builtin_amdgcn_wmma_f32_16x16x32_bf16(
                    false, a[t], false, b[g], (short)0, acc[g][t], false, false);
    }

    // ---- epilogue: gates + state update ----
    // C/D layout: element i, lanes 0-15 -> M=i, N=lane; lanes 16-31 -> M=i+8, N=lane-16
    const int unit = u_tile * 16 + (lane & 15);
    const float bi = bias[unit];
    const float bf = bias[UNITS     + unit];
    const float bg = bias[2 * UNITS + unit];
    const float bo = bias[3 * UNITS + unit];
    float* out_h = out;
    float* out_c = out + (size_t)batch * UNITS;

#pragma unroll
    for (int t = 0; t < MTILES; ++t) {
        const int r0 = rowBase + t * 16 + ((lane & 16) ? 8 : 0);
#pragma unroll
        for (int i = 0; i < 8; ++i) {
            size_t idx = (size_t)(r0 + i) * UNITS + unit;
            float ig = sigm(acc[0][t][i] + bi);
            float fg = sigm(acc[1][t][i] + bf);
            float gg = tanhf(acc[2][t][i] + bg);
            float og = sigm(acc[3][t][i] + bo);
            float cv = c[idx];
            float cn = fg * cv + ig * gg;
            float hn = og * tanhf(cn);
            out_h[idx] = hn;
            out_c[idx] = cn;
        }
    }
}

extern "C" void kernel_launch(void* const* d_in, const int* in_sizes, int n_in,
                              void* d_out, int out_size, void* d_ws, size_t ws_size,
                              hipStream_t stream) {
    const float* x    = (const float*)d_in[0];
    const float* h    = (const float*)d_in[1];
    const float* c    = (const float*)d_in[2];
    const float* W    = (const float*)d_in[3];
    const float* U    = (const float*)d_in[4];
    const float* bias = (const float*)d_in[5];
    float*  out   = (float*)d_out;
    __bf16* wpack = (__bf16*)d_ws;                 // 512*1024*2 B = 1 MB

    const int batch = in_sizes[0] / IN_DIM;        // 65536

    // 1) repack weights fp32 -> bf16 B-fragments (L2-resident, rebuilt every call)
    int packThreads = NTILES * KSTEPS * 32;        // 32768
    pack_weights_bf16<<<(packThreads + 255) / 256, 256, 0, stream>>>(W, U, wpack);

    // 2) fused GEMM + LSTM gates: 64 rows x 128 units per block
    dim3 grid(batch / ROWS_PER_BLOCK, 2);
    lstm_wmma_bf16<<<grid, 256, 0, stream>>>(x, h, c, bias, wpack, out, batch);
}
